// JordanLeechMoE_65317862637744
// MI455X (gfx1250) — matrix operand order repaired and checked by reading, not compiled, not run on hardware
//
#include <hip/hip_runtime.h>
#include <hip/hip_bf16.h>

// ---------------------------------------------------------------------------
// MoE (top-3 of 24 experts, fixed Egyptian combine weights) for MI455X gfx1250.
//
// Pipeline per call (all stream-ordered, graph-capture safe):
//   0. memset counts + out
//   1. cvt: x, w1, w2 fp32 -> bf16 (one pass, ~603 MB @ 23.3 TB/s ~= 26 us)
//   2. gate_route: 1 wave/token, top-3 of 24, atomic ticketing
//   3. scan: 24-entry prefix sum -> packed h segment offsets (3T rows exactly)
//   4. ffn1: h = relu(gather(xb) @ w1b[e] + b1)   [bf16 WMMA 16x16x32]
//   5. ffn2: out += eg_k * (h @ w2b[e] + b2)      [bf16 WMMA + f32 atomics]
//
// GEMM hot loops: double-buffered LDS tiles filled with
// global_load_async_to_lds_b128 (ASYNCcnt), 4 async copies per wave per tile,
// s_wait_asynccnt 0x4 overlapping next-tile copies with current-tile WMMAs.
// Buffer selection is a runtime pointer so there is a single compute path
// (no accumulator v_mov copies, no WMMA->VALU hazard nops).
// ---------------------------------------------------------------------------

typedef __bf16 bf16_t;
typedef bf16_t v16bf __attribute__((ext_vector_type(16)));
typedef bf16_t v8bf  __attribute__((ext_vector_type(8)));
typedef float  v8f   __attribute__((ext_vector_type(8)));

constexpr int E  = 24;
constexpr int D  = 1024;
constexpr int F  = 2048;
constexpr int T  = 4096;          // B*L tokens
constexpr int BM = 128;           // token tile
constexpr int BN = 128;           // N tile
constexpr int KT = 32;            // K tile == WMMA bf16 K
constexpr int AS_STRIDE = 40;     // bf16 elems; 80B pitch (16B aligned, bank-skewed)
constexpr int BS_STRIDE = 136;    // bf16 elems; 272B pitch

// Workspace layout (bytes)
constexpr size_t WS_H   = size_t(1) << 20;                    // h: 3T x F bf16
constexpr size_t WS_XB  = WS_H  + size_t(3) * T * F * 2;      // xb: T x D bf16
constexpr size_t WS_W1B = WS_XB + size_t(T) * D * 2;          // w1b: E x D x F bf16
constexpr size_t WS_W2B = WS_W1B + size_t(E) * D * F * 2;     // w2b: E x F x D bf16

typedef const unsigned short (*tileA_t)[AS_STRIDE];
typedef const unsigned short (*tileB_t)[BS_STRIDE];

__device__ __forceinline__ unsigned short f2bf(float f) {
  union { float f; unsigned u; } cv; cv.f = f;
  unsigned u = cv.u + 0x7FFFu + ((cv.u >> 16) & 1u);   // round-to-nearest-even
  return (unsigned short)(u >> 16);
}

// CDNA5 async copy: 16B global -> LDS per lane, tracked by ASYNCcnt.
__device__ __forceinline__ void async_copy16(unsigned lds_off,
                                             unsigned long long gaddr) {
  asm volatile("global_load_async_to_lds_b128 %0, %1, off"
               :
               : "v"(lds_off), "v"(gaddr)
               : "memory");
}
#define WAIT_ASYNC_LE4() asm volatile("s_wait_asynccnt 0x4" ::: "memory")
#define WAIT_ASYNC_0()   asm volatile("s_wait_asynccnt 0x0" ::: "memory")

__device__ __forceinline__ unsigned lds_off_of(const void* p) {
  return (unsigned)(uintptr_t)p;   // LDS aperture: low 32 bits = LDS offset
}

// ---------------------------------------------------------------------------
// fp32 -> bf16 pack, 8 elems/thread, b128 in / b128 out. n % 8 == 0.
// ---------------------------------------------------------------------------
__global__ __launch_bounds__(256) void cvt_bf16_kernel(
    const float* __restrict__ src, unsigned short* __restrict__ dst, int n) {
  int i = (blockIdx.x * 256 + threadIdx.x) * 8;
  if (i >= n) return;
  float4 a = *(const float4*)(src + i);
  float4 b = *(const float4*)(src + i + 4);
  uint4 u;
  u.x = (unsigned)f2bf(a.x) | ((unsigned)f2bf(a.y) << 16);
  u.y = (unsigned)f2bf(a.z) | ((unsigned)f2bf(a.w) << 16);
  u.z = (unsigned)f2bf(b.x) | ((unsigned)f2bf(b.y) << 16);
  u.w = (unsigned)f2bf(b.z) | ((unsigned)f2bf(b.w) << 16);
  *(uint4*)(dst + i) = u;
}

// ---------------------------------------------------------------------------
// Gating + routing. One wave32 per token.
// ---------------------------------------------------------------------------
__global__ __launch_bounds__(256) void gate_route_kernel(
    const float* __restrict__ x, const float* __restrict__ gate_w,
    int* __restrict__ counts, int* __restrict__ tlist, float* __restrict__ wlist) {
  __shared__ float lg[8][E];
  const int lane = threadIdx.x & 31;
  const int wid  = threadIdx.x >> 5;
  const int t = blockIdx.x * 8 + wid;

  const float* xr = x + (size_t)t * D;
  float xv[32];
#pragma unroll
  for (int j = 0; j < 32; ++j) xv[j] = xr[j * 32 + lane];

  for (int e = 0; e < E; ++e) {
    const float* gr = gate_w + (size_t)e * D;
    float acc = 0.f;
#pragma unroll
    for (int j = 0; j < 32; ++j) acc += xv[j] * gr[j * 32 + lane];
#pragma unroll
    for (int m = 16; m >= 1; m >>= 1) acc += __shfl_xor(acc, m, 32);
    if (lane == 0) lg[wid][e] = acc;
  }

  if (lane == 0) {
    const float eg[3] = {0.5f, 1.0f / 3.0f, 1.0f / 6.0f};
    int u0 = -1, u1 = -1;
    for (int k = 0; k < 3; ++k) {
      int best = 0; float bv = -3.4e38f;
      for (int e = 0; e < E; ++e) {
        if (e == u0 || e == u1) continue;
        float v = lg[wid][e];
        if (v > bv) { bv = v; best = e; }
      }
      int tk = atomicAdd(&counts[best], 1);
      tlist[best * T + tk] = t;
      wlist[best * T + tk] = eg[k];
      if (k == 0) u0 = best; else if (k == 1) u1 = best;
    }
  }
}

__global__ void scan_counts_kernel(const int* __restrict__ counts,
                                   int* __restrict__ offsets) {
  if (threadIdx.x == 0) {
    int s = 0;
    for (int e = 0; e < E; ++e) { offsets[e] = s; s += counts[e]; }
    offsets[E] = s;  // == 3*T
  }
}

// ---------------------------------------------------------------------------
// WMMA tile compute: 8 waves as 4(M) x 2(N); wave owns 32x64 = 2x4 fragments.
// A-fragment per CDNA5 16-bit layout (lane row M=lane%16, K chunks
// {0..7,16..23}/{8..15,24..31}); B-fragment: lane col N=lane%16, same K chunks.
// As_/Bs_ are runtime tile pointers (double-buffer select without branching).
// ---------------------------------------------------------------------------
#define FFN_COMPUTE_STEP(As_, Bs_)                                             \
  {                                                                            \
    v16bf a[2], b[4];                                                          \
    const int ar = wm * 32 + (lane & 15);                                      \
    const int c0 = (lane < 16) ? 0 : 8;                                        \
    _Pragma("unroll") for (int mi = 0; mi < 2; ++mi) {                         \
      v8bf lo = *(const v8bf*)&As_[ar + mi * 16][c0];                          \
      v8bf hi = *(const v8bf*)&As_[ar + mi * 16][c0 + 16];                     \
      a[mi] = __builtin_shufflevector(lo, hi, 0, 1, 2, 3, 4, 5, 6, 7, 8, 9,    \
                                      10, 11, 12, 13, 14, 15);                 \
    }                                                                          \
    _Pragma("unroll") for (int ni = 0; ni < 4; ++ni)                           \
        _Pragma("unroll") for (int s = 0; s < 16; ++s)                         \
            b[ni][s] = __builtin_bit_cast(bf16_t,                              \
                Bs_[(s < 8) ? (c0 + s) : (c0 + 8 + s)]                         \
                   [wn * 64 + ni * 16 + (lane & 15)]);                         \
    _Pragma("unroll") for (int mi = 0; mi < 2; ++mi)                           \
        _Pragma("unroll") for (int ni = 0; ni < 4; ++ni)                       \
            acc[mi][ni] = __builtin_amdgcn_wmma_f32_16x16x32_bf16(             \
                false, a[mi], false, b[ni], (short)0, acc[mi][ni],             \
                false, false);                                                 \
  }

// ---------------------------------------------------------------------------
// ffn1: h = relu(gather(xb) @ w1b[e] + b1[e]) -> packed bf16 h buffer.
// Grid: (F/BN, T/BM, E); blocks past count[e] exit.
// ---------------------------------------------------------------------------
__global__ __launch_bounds__(256, 2) void ffn1_kernel(
    const unsigned short* __restrict__ xb, const unsigned short* __restrict__ w1b,
    const float* __restrict__ b1, const int* __restrict__ counts,
    const int* __restrict__ offsets, const int* __restrict__ tlist,
    unsigned short* __restrict__ h) {
  const int e   = blockIdx.z;
  const int cnt = counts[e];
  const int m0  = blockIdx.y * BM;
  if (m0 >= cnt) return;
  const int n0   = blockIdx.x * BN;
  const int hoff = offsets[e];

  __shared__ __align__(16) unsigned short As[2][BM][AS_STRIDE];
  __shared__ __align__(16) unsigned short Bs[2][KT][BS_STRIDE];
  __shared__ int tids[BM];

  const int tid  = threadIdx.x;
  const int lane = tid & 31, wid = tid >> 5;
  const int wm = wid & 3, wn = wid >> 2;

  for (int r = tid; r < BM; r += 256) {
    int gr = m0 + r; if (gr >= cnt) gr = cnt - 1;
    tids[r] = tlist[e * T + gr];
  }
  __syncthreads();

  // Fixed per-thread 16B copy chunks (2 for A, 2 for B) -> 4 async/wave/tile.
  // A tile: 128 rows x 32 bf16 = 512 chunks; chunk c: row=c>>2, sub=(c&3)*8.
  // B tile:  32 rows x 128 bf16 = 512 chunks; chunk c: kr=c>>4, cb=(c&15)*8.
  const unsigned short* w1e = w1b + (size_t)e * D * F;
  const int ca0 = tid, ca1 = tid + 256;
  const int ar0 = ca0 >> 2, as0 = (ca0 & 3) * 8;
  const int ar1 = ca1 >> 2, as1 = (ca1 & 3) * 8;
  const unsigned short* agp0 = xb + (size_t)tids[ar0] * D + as0;
  const unsigned short* agp1 = xb + (size_t)tids[ar1] * D + as1;
  const int bk0 = ca0 >> 4, bc0 = (ca0 & 15) * 8;
  const int bk1 = ca1 >> 4, bc1 = (ca1 & 15) * 8;
  const unsigned short* bgp0 = w1e + (size_t)bk0 * F + n0 + bc0;
  const unsigned short* bgp1 = w1e + (size_t)bk1 * F + n0 + bc1;

  auto issue = [&](int kb, int buf) {
    async_copy16(lds_off_of(&As[buf][ar0][as0]),
                 (unsigned long long)(uintptr_t)(agp0 + kb));
    async_copy16(lds_off_of(&As[buf][ar1][as1]),
                 (unsigned long long)(uintptr_t)(agp1 + kb));
    async_copy16(lds_off_of(&Bs[buf][bk0][bc0]),
                 (unsigned long long)(uintptr_t)(bgp0 + (size_t)kb * F));
    async_copy16(lds_off_of(&Bs[buf][bk1][bc1]),
                 (unsigned long long)(uintptr_t)(bgp1 + (size_t)kb * F));
  };

  v8f zero;
#pragma unroll
  for (int q = 0; q < 8; ++q) zero[q] = 0.0f;
  v8f acc[2][4];
#pragma unroll
  for (int mi = 0; mi < 2; ++mi)
#pragma unroll
    for (int ni = 0; ni < 4; ++ni) acc[mi][ni] = zero;

  constexpr int NKT = D / KT;
  issue(0, 0);
  for (int i = 0; i < NKT; ++i) {
    if (i + 1 < NKT) { issue((i + 1) * KT, (i + 1) & 1); WAIT_ASYNC_LE4(); }
    else             { WAIT_ASYNC_0(); }
    __syncthreads();
    tileA_t Asb = As[i & 1];
    tileB_t Bsb = Bs[i & 1];
    FFN_COMPUTE_STEP(Asb, Bsb);
    __syncthreads();
  }

  // Epilogue: bias + relu, store bf16 h rows into this expert's segment.
  const float* b1e = b1 + (size_t)e * F;
#pragma unroll
  for (int ni = 0; ni < 4; ++ni) {
    const int col = n0 + wn * 64 + ni * 16 + (lane & 15);
    const float bb = b1e[col];
#pragma unroll
    for (int mi = 0; mi < 2; ++mi) {
#pragma unroll
      for (int r = 0; r < 8; ++r) {
        int row = wm * 32 + mi * 16 + r + ((lane >> 4) << 3);
        if (m0 + row < cnt) {
          float v = acc[mi][ni][r] + bb;
          h[(size_t)(hoff + m0 + row) * F + col] = f2bf(v > 0.f ? v : 0.f);
        }
      }
    }
  }
}

// ---------------------------------------------------------------------------
// ffn2: out[token] += eg_k * (h @ w2b[e] + b2[e])  (f32 global atomics).
// Grid: (D/BN, T/BM, E).
// ---------------------------------------------------------------------------
__global__ __launch_bounds__(256, 2) void ffn2_kernel(
    const unsigned short* __restrict__ h, const unsigned short* __restrict__ w2b,
    const float* __restrict__ b2, const int* __restrict__ counts,
    const int* __restrict__ offsets, const int* __restrict__ tlist,
    const float* __restrict__ wlist, float* __restrict__ out) {
  const int e   = blockIdx.z;
  const int cnt = counts[e];
  const int m0  = blockIdx.y * BM;
  if (m0 >= cnt) return;
  const int n0   = blockIdx.x * BN;
  const int hoff = offsets[e];

  __shared__ __align__(16) unsigned short As[2][BM][AS_STRIDE];
  __shared__ __align__(16) unsigned short Bs[2][KT][BS_STRIDE];
  __shared__ int   tids[BM];
  __shared__ float wls[BM];

  const int tid  = threadIdx.x;
  const int lane = tid & 31, wid = tid >> 5;
  const int wm = wid & 3, wn = wid >> 2;

  for (int r = tid; r < BM; r += 256) {
    int gr = m0 + r; int src = (gr < cnt) ? gr : (cnt - 1);
    tids[r] = tlist[e * T + src];
    wls[r]  = (gr < cnt) ? wlist[e * T + src] : 0.f;
  }
  __syncthreads();

  const unsigned short* w2e  = w2b + (size_t)e * F * D;
  const unsigned short* hseg = h + (size_t)hoff * F;

  const int ca0 = tid, ca1 = tid + 256;
  const int ar0 = ca0 >> 2, as0 = (ca0 & 3) * 8;
  const int ar1 = ca1 >> 2, as1 = (ca1 & 3) * 8;
  const int hr0 = (m0 + ar0 < cnt) ? (m0 + ar0) : (cnt - 1);
  const int hr1 = (m0 + ar1 < cnt) ? (m0 + ar1) : (cnt - 1);
  const unsigned short* agp0 = hseg + (size_t)hr0 * F + as0;
  const unsigned short* agp1 = hseg + (size_t)hr1 * F + as1;
  const int bk0 = ca0 >> 4, bc0 = (ca0 & 15) * 8;
  const int bk1 = ca1 >> 4, bc1 = (ca1 & 15) * 8;
  const unsigned short* bgp0 = w2e + (size_t)bk0 * D + n0 + bc0;
  const unsigned short* bgp1 = w2e + (size_t)bk1 * D + n0 + bc1;

  auto issue = [&](int kb, int buf) {
    async_copy16(lds_off_of(&As[buf][ar0][as0]),
                 (unsigned long long)(uintptr_t)(agp0 + kb));
    async_copy16(lds_off_of(&As[buf][ar1][as1]),
                 (unsigned long long)(uintptr_t)(agp1 + kb));
    async_copy16(lds_off_of(&Bs[buf][bk0][bc0]),
                 (unsigned long long)(uintptr_t)(bgp0 + (size_t)kb * D));
    async_copy16(lds_off_of(&Bs[buf][bk1][bc1]),
                 (unsigned long long)(uintptr_t)(bgp1 + (size_t)kb * D));
  };

  v8f zero;
#pragma unroll
  for (int q = 0; q < 8; ++q) zero[q] = 0.0f;
  v8f acc[2][4];
#pragma unroll
  for (int mi = 0; mi < 2; ++mi)
#pragma unroll
    for (int ni = 0; ni < 4; ++ni) acc[mi][ni] = zero;

  constexpr int NKT = F / KT;
  issue(0, 0);
  for (int i = 0; i < NKT; ++i) {
    if (i + 1 < NKT) { issue((i + 1) * KT, (i + 1) & 1); WAIT_ASYNC_LE4(); }
    else             { WAIT_ASYNC_0(); }
    __syncthreads();
    tileA_t Asb = As[i & 1];
    tileB_t Bsb = Bs[i & 1];
    FFN_COMPUTE_STEP(Asb, Bsb);
    __syncthreads();
  }

  // Epilogue: bias, Egyptian weight, scatter-accumulate into out.
  const float* b2e = b2 + (size_t)e * D;
#pragma unroll
  for (int ni = 0; ni < 4; ++ni) {
    const int col = n0 + wn * 64 + ni * 16 + (lane & 15);
    const float bb = b2e[col];
#pragma unroll
    for (int mi = 0; mi < 2; ++mi) {
#pragma unroll
      for (int r = 0; r < 8; ++r) {
        int row = wm * 32 + mi * 16 + r + ((lane >> 4) << 3);
        if (m0 + row < cnt) {
          float v = wls[row] * (acc[mi][ni][r] + bb);
          atomicAdd(&out[(size_t)tids[row] * D + col], v);
        }
      }
    }
  }
}

// ---------------------------------------------------------------------------
// Host launcher. Workspace: 1 MB header (counts/offsets/tlist/wlist), then
// h (50.3 MB), xb (8.4 MB), w1b (100.7 MB), w2b (100.7 MB)  ~= 261 MB.
// ---------------------------------------------------------------------------
extern "C" void kernel_launch(void* const* d_in, const int* in_sizes, int n_in,
                              void* d_out, int out_size, void* d_ws, size_t ws_size,
                              hipStream_t stream) {
  const float* x      = (const float*)d_in[0];
  const float* gate_w = (const float*)d_in[1];
  const float* w1     = (const float*)d_in[2];
  const float* b1     = (const float*)d_in[3];
  const float* w2     = (const float*)d_in[4];
  const float* b2     = (const float*)d_in[5];
  float* out = (float*)d_out;

  char* ws = (char*)d_ws;
  int*   counts  = (int*)ws;
  int*   offsets = counts + 64;
  int*   tlist   = counts + 128;
  float* wlist   = (float*)(counts + 128 + E * T);
  unsigned short* h   = (unsigned short*)(ws + WS_H);
  unsigned short* xb  = (unsigned short*)(ws + WS_XB);
  unsigned short* w1b = (unsigned short*)(ws + WS_W1B);
  unsigned short* w2b = (unsigned short*)(ws + WS_W2B);

  (void)in_sizes; (void)n_in; (void)out_size; (void)ws_size;

  hipMemsetAsync(d_ws, 0, 4096, stream);                        // counts/offsets
  hipMemsetAsync(d_out, 0, (size_t)T * D * sizeof(float), stream);

  // One-time (per call) fp32 -> bf16 conversions.
  {
    int n1 = E * D * F;                 // 50,331,648
    int n2 = E * F * D;
    int nx = T * D;                     // 4,194,304
    cvt_bf16_kernel<<<n1 / (256 * 8), 256, 0, stream>>>(w1, w1b, n1);
    cvt_bf16_kernel<<<n2 / (256 * 8), 256, 0, stream>>>(w2, w2b, n2);
    cvt_bf16_kernel<<<nx / (256 * 8), 256, 0, stream>>>(x, xb, nx);
  }

  gate_route_kernel<<<T / 8, 256, 0, stream>>>(x, gate_w, counts, tlist, wlist);
  scan_counts_kernel<<<1, 32, 0, stream>>>(counts, offsets);
  ffn1_kernel<<<dim3(F / BN, T / BM, E), 256, 0, stream>>>(
      xb, w1b, b1, counts, offsets, tlist, h);
  ffn2_kernel<<<dim3(D / BN, T / BM, E), 256, 0, stream>>>(
      h, w2b, b2, counts, offsets, tlist, wlist, out);
}
